// MultiLabelRankingLossWithSoftTarget_32418413150579
// MI455X (gfx1250) — compile-verified, matrix operand order checked by reading, not run on hardware
//
#include <hip/hip_runtime.h>
#include <hip/hip_bf16.h>

// MultiLabelRankingLossWithSoftTarget, MARGIN = 1.0, C = 64.
// Since probs = sigmoid(logits) in (0,1), |p_x - p_y| < 1 = MARGIN, so the
// hinge max(0, margin - diff) is ALWAYS the linear branch:
//   loss_sum = sum_x (1-p_x)*rowsum_x(M) + sum_y p_y*colsum_y(M)
//   n_valid  = sum_x rowsum_x(M),  M_{xy} = [t_x > t_y]
// Row/col sums of the 64x64 0/1 mask are computed on the matrix pipes via
// v_wmma_f32_16x16x32_f16 against an all-ones B operand (exact: 0/1 in f16,
// counts <= 64 in f32 accumulators).

typedef _Float16 v16h __attribute__((ext_vector_type(16)));
typedef float    v8f  __attribute__((ext_vector_type(8)));

#define C_DIM 64
#define WPB   8   // waves (samples) per 256-thread block

__global__ __launch_bounds__(256, 2)
void mlrl_stage1(const float* __restrict__ logits,
                 const float* __restrict__ target,
                 float* __restrict__ block_sums, int nB)
{
    __shared__ float s_t[WPB][C_DIM];
    __shared__ float s_p[WPB][C_DIM];
    __shared__ float s_ws[WPB];

    const int lane = threadIdx.x & 31;
    const int wv   = threadIdx.x >> 5;
    const int s    = blockIdx.x * WPB + wv;
    const bool active = (s < nB);

    if (active) {
        const float* lg = logits + (size_t)s * C_DIM;
        const float* tg = target + (size_t)s * C_DIM;
        if (lane == 0 && (s + WPB) < nB) {
            // gfx1250: global_prefetch_b8 for the next sample group
            __builtin_prefetch(logits + (size_t)(s + WPB) * C_DIM, 0, 0);
            __builtin_prefetch(target + (size_t)(s + WPB) * C_DIM, 0, 0);
        }
        const float x0 = lg[lane], x1 = lg[lane + 32];
        s_t[wv][lane]      = tg[lane];
        s_t[wv][lane + 32] = tg[lane + 32];
        s_p[wv][lane]      = 1.0f / (1.0f + __expf(-x0));   // v_exp_f32
        s_p[wv][lane + 32] = 1.0f / (1.0f + __expf(-x1));
    }
    __syncthreads();

    float per_sample = 0.0f;
    if (active) {   // wave-uniform branch: EXEC all-1s inside (WMMA requirement)
        const float* t = s_t[wv];
        const float* p = s_p[wv];
        const int hi = lane >> 4;   // half-wave selects K sub-block (ISA A layout)
        const int rl = lane & 15;   // row within 16x32 A tile

        v16h bones;
        #pragma unroll
        for (int j = 0; j < 16; ++j) bones[j] = (_Float16)1.0f;

        float tr[4];
        #pragma unroll
        for (int i = 0; i < 4; ++i) tr[i] = t[16 * i + rl];

        const v8f vzero = {0.f, 0.f, 0.f, 0.f, 0.f, 0.f, 0.f, 0.f};
        v8f crow[4], ccol[4];
        #pragma unroll
        for (int i = 0; i < 4; ++i) { crow[i] = vzero; ccol[i] = vzero; }

        // 16-bit A 16x32 layout: lane holds K = (j<8?0:16) + 8*hi + (j&7)
        // within the 32-wide K half h. K-operand values shared by all 4 tile
        // rows and both directions -> hoist.
        #pragma unroll
        for (int h = 0; h < 2; ++h) {
            float kv[16];
            const int b0 = 32 * h + 8 * hi;
            const int b1 = 32 * h + 16 + 8 * hi;
            #pragma unroll
            for (int j = 0; j < 8; ++j) { kv[j] = t[b0 + j]; kv[8 + j] = t[b1 + j]; }

            #pragma unroll
            for (int i = 0; i < 4; ++i) {
                v16h ar, ac;
                #pragma unroll
                for (int j = 0; j < 16; ++j) {
                    ar[j] = (tr[i] > kv[j]) ? (_Float16)1.0f : (_Float16)0.0f; // M
                    ac[j] = (kv[j] > tr[i]) ? (_Float16)1.0f : (_Float16)0.0f; // M^T
                }
                crow[i] = __builtin_amdgcn_wmma_f32_16x16x32_f16(
                    false, ar, false, bones, (short)0, crow[i], false, false);
                ccol[i] = __builtin_amdgcn_wmma_f32_16x16x32_f16(
                    false, ac, false, bones, (short)0, ccol[i], false, false);
            }
        }

        // D layout: (lane, vgpr j) -> row M = 16*i + 8*hi + j, replicated over
        // 16 columns N => every true value is counted 16x across the wave.
        float accL = 0.0f, accN = 0.0f;
        #pragma unroll
        for (int i = 0; i < 4; ++i) {
            #pragma unroll
            for (int j = 0; j < 8; ++j) {
                const int   M  = 16 * i + 8 * hi + j;
                const float pm = p[M];
                const float rs = crow[i][j];
                const float cs = ccol[i][j];
                accL += (1.0f - pm) * rs + pm * cs;
                accN += rs;
            }
        }
        #pragma unroll
        for (int off = 16; off > 0; off >>= 1) {
            accL += __shfl_xor(accL, off, 32);
            accN += __shfl_xor(accN, off, 32);
        }
        // accL = 16*loss_sum, accN = 16*n_valid  =>  loss/max(n_valid,1):
        per_sample = accL / fmaxf(accN, 16.0f);
    }
    if (lane == 0) s_ws[wv] = per_sample;
    __syncthreads();
    if (threadIdx.x == 0) {
        float b = 0.0f;
        #pragma unroll
        for (int k = 0; k < WPB; ++k) b += s_ws[k];
        block_sums[blockIdx.x] = b;
    }
}

__global__ __launch_bounds__(256)
void mlrl_stage2(const float* __restrict__ block_sums,
                 float* __restrict__ out, int nblocks, int nB)
{
    __shared__ float sm[256];
    float v = 0.0f;
    for (int i = threadIdx.x; i < nblocks; i += 256) v += block_sums[i];
    sm[threadIdx.x] = v;
    __syncthreads();
    for (int st = 128; st > 0; st >>= 1) {
        if ((int)threadIdx.x < st) sm[threadIdx.x] += sm[threadIdx.x + st];
        __syncthreads();
    }
    if (threadIdx.x == 0) out[0] = sm[0] / (float)nB;
}

extern "C" void kernel_launch(void* const* d_in, const int* in_sizes, int n_in,
                              void* d_out, int out_size, void* d_ws, size_t ws_size,
                              hipStream_t stream) {
    const float* logits = (const float*)d_in[0];
    const float* target = (const float*)d_in[1];
    const int nB = in_sizes[0] / C_DIM;          // 16384
    const int nblocks = (nB + WPB - 1) / WPB;    // 2048 -> 8 KB of d_ws
    float* bsum = (float*)d_ws;

    mlrl_stage1<<<nblocks, 256, 0, stream>>>(logits, target, bsum, nB);
    mlrl_stage2<<<1, 256, 0, stream>>>(bsum, (float*)d_out, nblocks, nB);
}
